// Decoder_3788161155715
// MI455X (gfx1250) — matrix operand order repaired and checked by reading, not compiled, not run on hardware
//
#include <hip/hip_runtime.h>

#define LL 2
#define BB 128
#define TT 512
#define DD 64
#define FF 512
#define NWG 32

#define LDS_ROW 520                 // 512 + 8 pad: lane stride 260 dwords -> bank-conflict free
#define LDS_GATE (16 * LDS_ROW)     // one gate block: 16 rows (byte stride 16640 < 64K: DS imm)
#define LDS_MAT  (4 * LDS_GATE)     // one weight matrix: 4 gates
#define HN (BB * FF)                // one h buffer, elements

typedef __attribute__((ext_vector_type(16))) __bf16 v16bf;
typedef __attribute__((ext_vector_type(8)))  __bf16 v8bf;
typedef __attribute__((ext_vector_type(8)))  float  v8f;

union ABOp { v16bf v; v8bf h[2]; };

__device__ __forceinline__ unsigned short f2bf(float x) {
  unsigned int u = __float_as_uint(x);
  u += 0x7FFFu + ((u >> 16) & 1u);            // round-to-nearest-even
  return (unsigned short)(u >> 16);
}

__device__ __forceinline__ float sigm(float x) {
  return 1.0f / (1.0f + __expf(-x));
}
__device__ __forceinline__ float tanh_fast(float x) {
  return 1.0f - 2.0f / (__expf(2.0f * x) + 1.0f);
}

// ---------------- prologue kernels ----------------

__global__ void k_init_bar(int* bar) {
  if (threadIdx.x == 0) { bar[0] = 0; bar[1] = 0; }
}

__global__ void k_f32_to_bf16(const float* __restrict__ src,
                              unsigned short* __restrict__ dst, int n) {
  int i = blockIdx.x * blockDim.x + threadIdx.x;
  if (i < n) dst[i] = f2bf(src[i]);
}

// h_init[l,b,f] = x[l,b,:] . W_init[f,:] + b_init[f];  bf16 copies into h-region
__global__ void k_init_state(const float* __restrict__ x,
                             const float* __restrict__ Wi,
                             const float* __restrict__ bi,
                             float* __restrict__ hinit,
                             unsigned short* __restrict__ hreg) {
  int i = blockIdx.x * blockDim.x + threadIdx.x;
  if (i >= LL * BB * FF) return;
  int f = i % FF;
  int b = (i / FF) % BB;
  int l = i / (FF * BB);
  const float* xr = x + (size_t)(l * BB + b) * DD;
  const float* wr = Wi + (size_t)f * DD;
  float s = bi[f];
#pragma unroll 8
  for (int d = 0; d < DD; ++d) s += xr[d] * wr[d];
  hinit[i] = s;
  // h-region layout: [0]=last_feat, [1..2]=h0 buffers, [3..4]=h1 buffers
  size_t slot = (l == 0) ? 1 : 3;
  hreg[slot * HN + (size_t)b * FF + f] = f2bf(s);
}

// ---------------- persistent LSTM kernel ----------------

__device__ __forceinline__ void grid_barrier(int* __restrict__ bar) {
  __syncthreads();
  __threadfence();
  if (threadIdx.x == 0) {
    int* cnt = bar;
    int* gen = bar + 1;
    int g = __hip_atomic_load(gen, __ATOMIC_RELAXED, __HIP_MEMORY_SCOPE_AGENT);
    if (__hip_atomic_fetch_add(cnt, 1, __ATOMIC_ACQ_REL, __HIP_MEMORY_SCOPE_AGENT) == NWG - 1) {
      __hip_atomic_store(cnt, 0, __ATOMIC_RELAXED, __HIP_MEMORY_SCOPE_AGENT);
      __hip_atomic_fetch_add(gen, 1, __ATOMIC_RELEASE, __HIP_MEMORY_SCOPE_AGENT);
    } else {
      while (__hip_atomic_load(gen, __ATOMIC_ACQUIRE, __HIP_MEMORY_SCOPE_AGENT) == g) {
        __builtin_amdgcn_s_sleep(1);
      }
    }
  }
  __syncthreads();
  __threadfence();
}

// One LSTM layer step for one wave's 16x16 (batch x feature) tile.
// A operands from global (L2-resident h buffers), B operands from LDS weights.
// Each matrix gets its own pointer induction so all DS loads use immediate offsets.
__device__ __forceinline__ void lstm_layer_step(
    const __bf16* __restrict__ a_in,   // (BB,FF) bf16, global
    const __bf16* __restrict__ a_h,    // (BB,FF) bf16, global
    const __bf16* __restrict__ wls,    // LDS: Wih block  [gate][16 rows][LDS_ROW]
    const __bf16* __restrict__ whs,    // LDS: Whh block
    float bI, float bF, float bG, float bO,
    v8f& c,
    unsigned short* __restrict__ hout_bf,   // (BB,FF), global
    float* __restrict__ hout_f32,           // (BB,FF) slice of d_out, or nullptr
    int b_base, int f_base, int lane)
{
  const int lane15 = lane & 15;
  const int hi     = lane >> 4;

  v8f acc_i = {0.f,0.f,0.f,0.f,0.f,0.f,0.f,0.f};
  v8f acc_f = {0.f,0.f,0.f,0.f,0.f,0.f,0.f,0.f};
  v8f acc_g = {0.f,0.f,0.f,0.f,0.f,0.f,0.f,0.f};
  v8f acc_o = {0.f,0.f,0.f,0.f,0.f,0.f,0.f,0.f};

  // A rows: m = lane&15; lane holds 16 contiguous K, lanes 16-31 at K+16
  const __bf16* pin = a_in + (size_t)(b_base + lane15) * FF + hi * 16;
  const __bf16* pah = a_h  + (size_t)(b_base + lane15) * FF + hi * 16;
  // B cols: n = lane&15; lane n holds 16 contiguous K, lanes 16-31 at K+16
  const __bf16* pwi = wls + lane15 * LDS_ROW + hi * 16;
  const __bf16* pwh = whs + lane15 * LDS_ROW + hi * 16;

#pragma unroll 4
  for (int kb = 0; kb < FF; kb += 32) {
    ABOp aIn, aH;
    aIn.h[0] = *(const v8bf*)(pin);
    aIn.h[1] = *(const v8bf*)(pin + 8);
    aH.h[0]  = *(const v8bf*)(pah);
    aH.h[1]  = *(const v8bf*)(pah + 8);

    ABOp wi_i, wh_i, wi_f, wh_f, wi_g, wh_g, wi_o, wh_o;
    wi_i.h[0] = *(const v8bf*)(pwi + 0 * LDS_GATE);
    wi_i.h[1] = *(const v8bf*)(pwi + 0 * LDS_GATE + 8);
    wh_i.h[0] = *(const v8bf*)(pwh + 0 * LDS_GATE);
    wh_i.h[1] = *(const v8bf*)(pwh + 0 * LDS_GATE + 8);
    acc_i = __builtin_amdgcn_wmma_f32_16x16x32_bf16(false, aIn.v, false, wi_i.v, (short)0, acc_i, false, false);
    acc_i = __builtin_amdgcn_wmma_f32_16x16x32_bf16(false, aH.v,  false, wh_i.v, (short)0, acc_i, false, false);

    wi_f.h[0] = *(const v8bf*)(pwi + 1 * LDS_GATE);
    wi_f.h[1] = *(const v8bf*)(pwi + 1 * LDS_GATE + 8);
    wh_f.h[0] = *(const v8bf*)(pwh + 1 * LDS_GATE);
    wh_f.h[1] = *(const v8bf*)(pwh + 1 * LDS_GATE + 8);
    acc_f = __builtin_amdgcn_wmma_f32_16x16x32_bf16(false, aIn.v, false, wi_f.v, (short)0, acc_f, false, false);
    acc_f = __builtin_amdgcn_wmma_f32_16x16x32_bf16(false, aH.v,  false, wh_f.v, (short)0, acc_f, false, false);

    wi_g.h[0] = *(const v8bf*)(pwi + 2 * LDS_GATE);
    wi_g.h[1] = *(const v8bf*)(pwi + 2 * LDS_GATE + 8);
    wh_g.h[0] = *(const v8bf*)(pwh + 2 * LDS_GATE);
    wh_g.h[1] = *(const v8bf*)(pwh + 2 * LDS_GATE + 8);
    acc_g = __builtin_amdgcn_wmma_f32_16x16x32_bf16(false, aIn.v, false, wi_g.v, (short)0, acc_g, false, false);
    acc_g = __builtin_amdgcn_wmma_f32_16x16x32_bf16(false, aH.v,  false, wh_g.v, (short)0, acc_g, false, false);

    wi_o.h[0] = *(const v8bf*)(pwi + 3 * LDS_GATE);
    wi_o.h[1] = *(const v8bf*)(pwi + 3 * LDS_GATE + 8);
    wh_o.h[0] = *(const v8bf*)(pwh + 3 * LDS_GATE);
    wh_o.h[1] = *(const v8bf*)(pwh + 3 * LDS_GATE + 8);
    acc_o = __builtin_amdgcn_wmma_f32_16x16x32_bf16(false, aIn.v, false, wi_o.v, (short)0, acc_o, false, false);
    acc_o = __builtin_amdgcn_wmma_f32_16x16x32_bf16(false, aH.v,  false, wh_o.v, (short)0, acc_o, false, false);

    pin += 32; pah += 32; pwi += 32; pwh += 32;
  }

  // C/D layout: lane col n = f_base + lane&15; VGPR r -> row m = r + 8*hi
  unsigned short* ob = hout_bf + (size_t)(b_base + 8 * hi) * FF + f_base + lane15;
  float* of = hout_f32 ? hout_f32 + (size_t)(b_base + 8 * hi) * FF + f_base + lane15 : nullptr;
#pragma unroll
  for (int r = 0; r < 8; ++r) {
    float iv = sigm(acc_i[r] + bI);
    float fv = sigm(acc_f[r] + bF);
    float gv = tanh_fast(acc_g[r] + bG);
    float ov = sigm(acc_o[r] + bO);
    float cn = fv * c[r] + iv * gv;
    c[r] = cn;
    float hn = ov * tanh_fast(cn);
    ob[(size_t)r * FF] = f2bf(hn);
    if (of) __builtin_nontemporal_store(hn, of + (size_t)r * FF);
  }
}

__global__ void __launch_bounds__(256)
k_lstm_persistent(const float* __restrict__ hinit,
                  const unsigned short* __restrict__ Wb0,  // bf16 bits: ih0
                  const unsigned short* __restrict__ Wb1,  // hh0
                  const unsigned short* __restrict__ Wb2,  // ih1
                  const unsigned short* __restrict__ Wb3,  // hh1
                  const float* __restrict__ b_ih0, const float* __restrict__ b_hh0,
                  const float* __restrict__ b_ih1, const float* __restrict__ b_hh1,
                  unsigned short* __restrict__ hreg,       // [5][BB*FF]: lf, h0x2, h1x2
                  float* __restrict__ out,
                  int* __restrict__ bar)
{
  extern __shared__ __bf16 smem[];   // [mat(4)][gate(4)][row(16)][LDS_ROW] = 260 KB

  const int tid    = threadIdx.x;
  const int lane   = tid & 31;
  const int b_tile = tid >> 5;         // 8 batch tiles: one per wave in the WG
  const int f_tile = blockIdx.x;       // 32 feature tiles: one per WG (shared weights!)
  const int b_base = b_tile * 16;
  const int f_base = f_tile * 16;
  const int lane15 = lane & 15;
  const int hi     = lane >> 4;
  const int ncol   = f_base + lane15;

  // ---- stage this WG's weight columns into LDS once (reused for all 512 steps) ----
  {
    const unsigned short* mats[4] = {Wb0, Wb1, Wb2, Wb3};
    const int mat  = tid >> 6;         // 0..3
    const int gate = (tid >> 4) & 3;   // 0..3
    const int row  = tid & 15;         // 0..15
    const __bf16* src = (const __bf16*)mats[mat] + (size_t)(gate * FF + f_base + row) * FF;
    __bf16* dst = smem + (size_t)tid * LDS_ROW;   // tid == (mat*4+gate)*16+row
#pragma unroll 8
    for (int k = 0; k < FF; k += 8)
      *(v8bf*)(dst + k) = *(const v8bf*)(src + k);
  }
  __syncthreads();

  // per-lane fused biases (one per gate, per layer) — constant across steps
  const float bI0 = b_ih0[0 * FF + ncol] + b_hh0[0 * FF + ncol];
  const float bF0 = b_ih0[1 * FF + ncol] + b_hh0[1 * FF + ncol];
  const float bG0 = b_ih0[2 * FF + ncol] + b_hh0[2 * FF + ncol];
  const float bO0 = b_ih0[3 * FF + ncol] + b_hh0[3 * FF + ncol];
  const float bI1 = b_ih1[0 * FF + ncol] + b_hh1[0 * FF + ncol];
  const float bF1 = b_ih1[1 * FF + ncol] + b_hh1[1 * FF + ncol];
  const float bG1 = b_ih1[2 * FF + ncol] + b_hh1[2 * FF + ncol];
  const float bO1 = b_ih1[3 * FF + ncol] + b_hh1[3 * FF + ncol];

  // cell state lives in registers for all 512 steps (c_init == h_init)
  v8f c0, c1;
#pragma unroll
  for (int r = 0; r < 8; ++r) {
    int row = b_base + r + 8 * hi;
    c0[r] = hinit[(size_t)(0 * BB + row) * FF + ncol];
    c1[r] = hinit[(size_t)(1 * BB + row) * FF + ncol];
  }

  const __bf16* wih0_l = smem + 0 * LDS_MAT;
  const __bf16* whh0_l = smem + 1 * LDS_MAT;
  const __bf16* wih1_l = smem + 2 * LDS_MAT;
  const __bf16* whh1_l = smem + 3 * LDS_MAT;

  for (int t = 0; t < TT; ++t) {
    const size_t p = (size_t)(t & 1), q = p ^ 1;
    // all h-buffer accesses are integer offsets from ONE global base -> global_load/store
    // slots: 0 = last_feat, 1+p = h0[p], 3+p = h1[p]
    const size_t in0_off = (t == 0) ? 0 : (3 + p) * (size_t)HN;

    // layer 0: input = prev h1 (last_feat at t==0); hidden = prev h0
    lstm_layer_step((const __bf16*)(hreg + in0_off),
                    (const __bf16*)(hreg + (1 + p) * (size_t)HN),
                    wih0_l, whh0_l,
                    bI0, bF0, bG0, bO0, c0,
                    hreg + (1 + q) * (size_t)HN, nullptr, b_base, f_base, lane);
    grid_barrier(bar);

    // layer 1: input = current h0; hidden = prev h1; h1 -> out[t]
    lstm_layer_step((const __bf16*)(hreg + (1 + q) * (size_t)HN),
                    (const __bf16*)(hreg + (3 + p) * (size_t)HN),
                    wih1_l, whh1_l,
                    bI1, bF1, bG1, bO1, c1,
                    hreg + (3 + q) * (size_t)HN,
                    out + (size_t)t * BB * FF, b_base, f_base, lane);
    grid_barrier(bar);
  }
}

// ---------------- host launcher ----------------

extern "C" void kernel_launch(void* const* d_in, const int* in_sizes, int n_in,
                              void* d_out, int out_size, void* d_ws, size_t ws_size,
                              hipStream_t stream) {
  (void)in_sizes; (void)n_in; (void)out_size; (void)ws_size;
  const float* x         = (const float*)d_in[0];
  const float* last_feat = (const float*)d_in[1];
  const float* W_init    = (const float*)d_in[2];
  const float* b_init    = (const float*)d_in[3];
  const float* W_ih0     = (const float*)d_in[4];
  const float* W_hh0     = (const float*)d_in[5];
  const float* b_ih0     = (const float*)d_in[6];
  const float* b_hh0     = (const float*)d_in[7];
  const float* W_ih1     = (const float*)d_in[8];
  const float* W_hh1     = (const float*)d_in[9];
  const float* b_ih1     = (const float*)d_in[10];
  const float* b_hh1     = (const float*)d_in[11];
  // W_out / b_out (d_in[12..13]) are dead code in the reference

  char* ws = (char*)d_ws;
  size_t off = 0;
  auto carve = [&](size_t bytes) -> char* {
    char* p = ws + off;
    off += (bytes + 255) & ~(size_t)255;
    return p;
  };
  const size_t WBYTES = (size_t)4 * FF * FF * sizeof(unsigned short); // 2 MB each
  unsigned short* Wb_ih0 = (unsigned short*)carve(WBYTES);
  unsigned short* Wb_hh0 = (unsigned short*)carve(WBYTES);
  unsigned short* Wb_ih1 = (unsigned short*)carve(WBYTES);
  unsigned short* Wb_hh1 = (unsigned short*)carve(WBYTES);
  // one contiguous h-region: [0]=last_feat_bf, [1..2]=h0 dbl-buf, [3..4]=h1 dbl-buf
  unsigned short* hreg = (unsigned short*)carve((size_t)5 * HN * sizeof(unsigned short));
  float* hinit = (float*)carve((size_t)LL * BB * FF * sizeof(float));  // 512 KB
  int* bar     = (int*)carve(256);

  k_init_bar<<<1, 32, 0, stream>>>(bar);

  const int nW = 4 * FF * FF;
  k_f32_to_bf16<<<(nW + 255) / 256, 256, 0, stream>>>(W_ih0, Wb_ih0, nW);
  k_f32_to_bf16<<<(nW + 255) / 256, 256, 0, stream>>>(W_hh0, Wb_hh0, nW);
  k_f32_to_bf16<<<(nW + 255) / 256, 256, 0, stream>>>(W_ih1, Wb_ih1, nW);
  k_f32_to_bf16<<<(nW + 255) / 256, 256, 0, stream>>>(W_hh1, Wb_hh1, nW);
  const int nLF = BB * FF;
  k_f32_to_bf16<<<(nLF + 255) / 256, 256, 0, stream>>>(last_feat, hreg, nLF); // slot 0

  const int nI = LL * BB * FF;
  k_init_state<<<(nI + 255) / 256, 256, 0, stream>>>(x, W_init, b_init, hinit, hreg);

  const size_t ldsBytes = (size_t)4 * LDS_MAT * sizeof(__bf16);   // 260 KB < 320 KB/WGP
  k_lstm_persistent<<<NWG, 256, ldsBytes, stream>>>(
      hinit, Wb_ih0, Wb_hh0, Wb_ih1, Wb_hh1,
      b_ih0, b_hh0, b_ih1, b_hh1,
      hreg, (float*)d_out, bar);
}
